// MRNN_fixD_66726611911225
// MI455X (gfx1250) — matrix operand order repaired and checked by reading, not compiled, not run on hardware
//
#include <hip/hip_runtime.h>
#include <cstdint>
#include <cstddef>

// Problem dims (fixed by reference)
#define TDIM 4096
#define BDIM 64
#define FDIM 128
#define HDIM 512
#define ODIM 128
#define KTAP 32

typedef __attribute__((ext_vector_type(16))) __bf16        v16bf;
typedef __attribute__((ext_vector_type(8)))  float         v8f;
typedef __attribute__((ext_vector_type(4)))  unsigned int  u32x4;
typedef __attribute__((ext_vector_type(8)))  unsigned int  u32x8;
typedef __attribute__((ext_vector_type(4)))  float         f32x4;
typedef __attribute__((ext_vector_type(8)))  int           i32x8;
typedef __attribute__((ext_vector_type(4)))  int           i32x4;

// ---------- helpers ----------
__device__ __forceinline__ unsigned short f2bf(float x) {
  unsigned u = __builtin_bit_cast(unsigned, x);
  u += 0x7fffu + ((u >> 16) & 1u);
  return (unsigned short)(u >> 16);
}

// 16 contiguous bf16 at p  (B-fragment pattern: lanes 0-15 K=0..15, 16-31 K=16..31)
__device__ __forceinline__ v16bf ld_frag_contig(const unsigned short* p) {
  u32x4 a = *(const u32x4*)(p);
  u32x4 b = *(const u32x4*)(p + 8);
  u32x8 c;
  c[0]=a[0]; c[1]=a[1]; c[2]=a[2]; c[3]=a[3];
  c[4]=b[0]; c[5]=b[1]; c[6]=b[2]; c[7]=b[3];
  return __builtin_bit_cast(v16bf, c);
}

// 16 contiguous bf16 at p  (A-fragment pattern: lanes 0-15 K=0..15, 16-31 K=16..31)
__device__ __forceinline__ v16bf ld_frag_split(const unsigned short* p) {
  u32x4 a = *(const u32x4*)(p);
  u32x4 b = *(const u32x4*)(p + 8);
  u32x8 c;
  c[0]=a[0]; c[1]=a[1]; c[2]=a[2]; c[3]=a[3];
  c[4]=b[0]; c[5]=b[1]; c[6]=b[2]; c[7]=b[3];
  return __builtin_bit_cast(v16bf, c);
}

// ---------- kernel 0: ARFIMA weights  kern[j][f], j=0..K (kern[K]=1) ----------
__global__ void fracw_kernel(const float* __restrict__ bd, float* __restrict__ kern) {
  int f = threadIdx.x;                       // F = 128 threads
  float d = 0.5f / (1.0f + expf(-bd[f]));    // 0.5 * sigmoid
  kern[KTAP * FDIM + f] = 1.0f;              // pi_0
  float w = 1.0f;
  for (int i = 0; i < KTAP; ++i) {
    w = w * ((float)i - d) / (float)(i + 1);
    kern[(KTAP - 1 - i) * FDIM + f] = w;     // [pi_K ... pi_1]
  }
}

// ---------- kernel 1: f32 -> bf16 weight conversion ----------
__global__ void cvt_bf16_kernel(const float* __restrict__ src,
                                unsigned short* __restrict__ dst, int n) {
  int i = blockIdx.x * 256 + threadIdx.x;
  if (i < n) dst[i] = f2bf(src[i]);
}

// ---------- kernel 2: depthwise causal conv (frac diff), TDM-staged LDS ----------
#define TT 64
__global__ void __launch_bounds__(256)
fracdiff_kernel(const float* __restrict__ x,          // [T][B][F]
                const float* __restrict__ kern,       // [K+1][F]
                unsigned short* __restrict__ xf) {    // [T*B][F] bf16
  __shared__ float xs[(TT + KTAP) * FDIM];            // 96*128*4 = 48 KB at LDS offset 0
  const int b  = blockIdx.x % BDIM;
  const int tc = blockIdx.x / BDIM;
  const int t0 = tc * TT;

  const int rows        = (tc == 0) ? TT : (TT + KTAP);
  const unsigned ldsoff = (tc == 0) ? (unsigned)(KTAP * FDIM * 4) : 0u;
  const size_t grow     = (tc == 0) ? 0 : (size_t)(t0 - KTAP);

  if (tc == 0) {  // zero-fill the negative-lag rows
    for (int i = threadIdx.x; i < KTAP * FDIM; i += 256) xs[i] = 0.0f;
  }

  if (threadIdx.x < 32) {  // wave 0 issues one 2-D TDM tile load: rows x F f32
    unsigned long long ga =
        (unsigned long long)(uintptr_t)(const void*)(x + (grow * BDIM + b) * FDIM);
    u32x4 g0;
    g0[0] = 1u;                                            // count=1, no gather
    g0[1] = ldsoff;                                        // lds_addr (xs @ LDS 0)
    g0[2] = (unsigned)(ga & 0xffffffffu);                  // global_addr[31:0]
    g0[3] = (unsigned)((ga >> 32) & 0x01ffffffu) | (2u << 30); // addr[56:32] | type=2
    i32x8 g1;
    g1[0] = (int)(2u << 16);                               // wg_mask=0, data_size=4B
    g1[1] = (int)(((unsigned)FDIM & 0xffffu) << 16);       // tensor_dim0[15:0]
    g1[2] = (int)(((unsigned)rows & 0xffffu) << 16);       // tensor_dim1[15:0]
    g1[3] = (int)(((unsigned)FDIM & 0xffffu) << 16);       // tile_dim0 = F
    g1[4] = (int)((unsigned)rows & 0xffffu);               // tile_dim1 = rows
    g1[5] = (int)(BDIM * FDIM);                            // tensor_dim0_stride[31:0]
    g1[6] = 0;                                             // stride hi / dim1_stride lo
    g1[7] = 0;
    i32x4 z4 = {0, 0, 0, 0};
    i32x8 z8 = {0, 0, 0, 0, 0, 0, 0, 0};
    __builtin_amdgcn_tensor_load_to_lds(g0, g1, z4, z4, z8, 0);
    __builtin_amdgcn_s_wait_tensorcnt(0);
  }
  __syncthreads();

  for (int o = threadIdx.x; o < TT * FDIM; o += 256) {
    int i = o >> 7;
    int f = o & (FDIM - 1);
    float acc = 0.0f;
#pragma unroll
    for (int j = 0; j <= KTAP; ++j)
      acc = fmaf(kern[j * FDIM + f], xs[(i + j) * FDIM + f], acc);
    xf[((size_t)(t0 + i) * BDIM + b) * FDIM + f] = f2bf(acc);
  }
}

// ---------- kernel 3: input projection GEMM (WMMA bf16) ----------
// xpT[t][h][b] = x_f[t*B+b, :] . W_ih[h, :] + b_ih[h] + b_hh[h]
__global__ void __launch_bounds__(256)
input_proj_kernel(const unsigned short* __restrict__ xf,    // [T*B][F] bf16
                  const unsigned short* __restrict__ Wihb,  // [H][F]  bf16
                  const float* __restrict__ bih,
                  const float* __restrict__ bhh,
                  float* __restrict__ xpT) {                // [T][H][B] f32
  const int lane = threadIdx.x & 31;
  const int wid  = threadIdx.x >> 5;       // 0..7 -> N group (4 N tiles each)
  const int hi   = lane >> 4;              // 0/1
  const int lmod = lane & 15;
  const int Mt   = blockIdx.x;             // 0..(T*B/16 - 1)

  v8f C[4];
#pragma unroll
  for (int j = 0; j < 4; ++j) {
    int n = (wid * 4 + j) * 16 + lmod;
    float bv = bih[n] + bhh[n];
#pragma unroll
    for (int r = 0; r < 8; ++r) C[j][r] = bv;
  }
#pragma unroll
  for (int kc = 0; kc < FDIM / 32; ++kc) {
    const unsigned short* pa = xf + ((size_t)Mt * 16 + lmod) * FDIM + kc * 32 + hi * 16;
    v16bf A = ld_frag_split(pa);
#pragma unroll
    for (int j = 0; j < 4; ++j) {
      int n = (wid * 4 + j) * 16 + lmod;
      const unsigned short* pb = Wihb + (size_t)n * FDIM + kc * 32 + hi * 16;
      v16bf Bf = ld_frag_contig(pb);
      C[j] = __builtin_amdgcn_wmma_f32_16x16x32_bf16(false, A, false, Bf,
                                                     (short)0, C[j], false, false);
    }
  }
  // store transposed: rows of this M-tile share one t (B=64 multiple of 16)
  const int t  = Mt >> 2;
  const int b0 = (Mt & 3) * 16 + hi * 8;
#pragma unroll
  for (int j = 0; j < 4; ++j) {
    int n = (wid * 4 + j) * 16 + lmod;
    float* p = xpT + ((size_t)t * HDIM + n) * BDIM + b0;
    f32x4 lo, hi4;
#pragma unroll
    for (int r = 0; r < 4; ++r) { lo[r] = C[j][r]; hi4[r] = C[j][r + 4]; }
    *(f32x4*)p       = lo;
    *(f32x4*)(p + 4) = hi4;
  }
}

// ---------- kernel 4: recurrence, single WGP ----------
// 32 waves = 32 N-tiles. 8 W_hh B-fragments resident in VGPRs, 8 streamed from
// L2 each step (bounds VGPR pressure at 8 waves/SIMD). h ping-pongs in LDS bf16.
__global__ void __launch_bounds__(1024)
rnn_recurrence_kernel(const float* __restrict__ xpT,            // [T][H][B]
                      const unsigned short* __restrict__ Whhb,  // [H][H] bf16
                      const unsigned short* __restrict__ Whob,  // [O][H] bf16
                      const float* __restrict__ bho,
                      float* __restrict__ out) {                // [B*O | B*H]
  extern __shared__ unsigned short hb[];                        // 2 * B*H bf16
  unsigned short* h0 = hb;
  unsigned short* h1 = hb + BDIM * HDIM;

  const int tid  = threadIdx.x;
  const int lane = tid & 31;
  const int wave = tid >> 5;        // 0..31 -> N tile
  const int hi   = lane >> 4;
  const int lmod = lane & 15;

  // h_0 = 0
  {
    unsigned int* z = (unsigned int*)h0;
    for (int i = tid; i < BDIM * HDIM / 2; i += 1024) z[i] = 0u;
  }
  __syncthreads();

  const int nt   = wave;
  const int ncol = nt * 16 + lmod;
  const unsigned short* Brow = Whhb + (size_t)ncol * HDIM + hi * 16;

  // Resident W_hh B-fragments for kc = 0..7 (64 VGPRs)
  v16bf Bfrag[8];
#pragma unroll
  for (int kc = 0; kc < 8; ++kc)
    Bfrag[kc] = ld_frag_contig(Brow + kc * 32);

  for (int t = 0; t < TDIM; ++t) {
    const unsigned short* hcur = (t & 1) ? h1 : h0;
    unsigned short*       hnxt = (t & 1) ? h0 : h1;

    // C init = xp_t tile (biases folded in xp); contiguous in transposed layout
    v8f C[4];
#pragma unroll
    for (int mt = 0; mt < 4; ++mt) {
      const float* p = xpT + ((size_t)t * HDIM + ncol) * BDIM + mt * 16 + hi * 8;
      f32x4 a = *(const f32x4*)p;
      f32x4 b = *(const f32x4*)(p + 4);
      v8f c;
#pragma unroll
      for (int r = 0; r < 4; ++r) { c[r] = a[r]; c[r + 4] = b[r]; }
      C[mt] = c;
    }
    if (t + 1 < TDIM)  // gfx1250 global_prefetch_b8 of next step's xp tile
      __builtin_prefetch(xpT + ((size_t)(t + 1) * HDIM + ncol) * BDIM, 0, 1);

    // kc 0..7: resident B
#pragma unroll 4
    for (int kc = 0; kc < 8; ++kc) {
#pragma unroll
      for (int mt = 0; mt < 4; ++mt) {
        const unsigned short* pa =
            hcur + (size_t)(mt * 16 + lmod) * HDIM + kc * 32 + hi * 16;
        v16bf A = ld_frag_split(pa);
        C[mt] = __builtin_amdgcn_wmma_f32_16x16x32_bf16(false, A, false, Bfrag[kc],
                                                        (short)0, C[mt], false, false);
      }
    }
    // kc 8..15: B streamed from L2 (W_hh bf16 = 512 KB, L2-resident)
#pragma unroll 4
    for (int kc = 8; kc < 16; ++kc) {
      v16bf Bs = ld_frag_contig(Brow + kc * 32);
#pragma unroll
      for (int mt = 0; mt < 4; ++mt) {
        const unsigned short* pa =
            hcur + (size_t)(mt * 16 + lmod) * HDIM + kc * 32 + hi * 16;
        v16bf A = ld_frag_split(pa);
        C[mt] = __builtin_amdgcn_wmma_f32_16x16x32_bf16(false, A, false, Bs,
                                                        (short)0, C[mt], false, false);
      }
    }

    // h_{t+1} = tanh(.) -> bf16 into the other LDS buffer
#pragma unroll
    for (int mt = 0; mt < 4; ++mt) {
#pragma unroll
      for (int r = 0; r < 8; ++r) {
        float v = tanhf(C[mt][r]);
        int m = mt * 16 + r + hi * 8;
        hnxt[(size_t)m * HDIM + ncol] = f2bf(v);
        if (t == TDIM - 1)  // hT output (f32), after the outputs region
          out[BDIM * ODIM + (size_t)m * HDIM + ncol] = v;
      }
    }
    __syncthreads();
  }

  // outputs = hT @ W_ho^T + b_ho   (final h lives in h0: T is even)
  const int mt = wave >> 3;          // 0..3
  const int ot = wave & 7;           // 0..7
  const int oc = ot * 16 + lmod;
  v8f C2;
  {
    float bv = bho[oc];
#pragma unroll
    for (int r = 0; r < 8; ++r) C2[r] = bv;
  }
#pragma unroll
  for (int kc = 0; kc < 16; ++kc) {
    const unsigned short* pa = h0 + (size_t)(mt * 16 + lmod) * HDIM + kc * 32 + hi * 16;
    v16bf A  = ld_frag_split(pa);
    const unsigned short* pb = Whob + (size_t)oc * HDIM + kc * 32 + hi * 16;
    v16bf Bf = ld_frag_contig(pb);
    C2 = __builtin_amdgcn_wmma_f32_16x16x32_bf16(false, A, false, Bf,
                                                 (short)0, C2, false, false);
  }
#pragma unroll
  for (int r = 0; r < 8; ++r) {
    int m = mt * 16 + r + hi * 8;
    out[(size_t)m * ODIM + oc] = C2[r];
  }
}

// ---------- workspace layout (bytes) ----------
#define WS_KERN  ((size_t)0)                               // (K+1)*F f32
#define WS_WIH   ((size_t)32768)                           // H*F bf16
#define WS_WHH   (WS_WIH + (size_t)HDIM * FDIM * 2)        // H*H bf16
#define WS_WHO   (WS_WHH + (size_t)HDIM * HDIM * 2)        // O*H bf16
#define WS_XF    ((size_t)819200)                          // T*B*F bf16
#define WS_XP    (WS_XF + (size_t)TDIM * BDIM * FDIM * 2)  // T*H*B f32

extern "C" void kernel_launch(void* const* d_in, const int* in_sizes, int n_in,
                              void* d_out, int out_size, void* d_ws, size_t ws_size,
                              hipStream_t stream) {
  const float* x   = (const float*)d_in[0];
  const float* bd  = (const float*)d_in[1];
  const float* Wih = (const float*)d_in[2];
  const float* bih = (const float*)d_in[3];
  const float* Whh = (const float*)d_in[4];
  const float* bhh = (const float*)d_in[5];
  const float* Who = (const float*)d_in[6];
  const float* bho = (const float*)d_in[7];
  float* out = (float*)d_out;

  char* ws = (char*)d_ws;
  float*          kern = (float*)(ws + WS_KERN);
  unsigned short* Wihb = (unsigned short*)(ws + WS_WIH);
  unsigned short* Whhb = (unsigned short*)(ws + WS_WHH);
  unsigned short* Whob = (unsigned short*)(ws + WS_WHO);
  unsigned short* xf   = (unsigned short*)(ws + WS_XF);
  float*          xpT  = (float*)(ws + WS_XP);

  // 0) ARFIMA filter taps
  fracw_kernel<<<1, FDIM, 0, stream>>>(bd, kern);

  // 1) weight conversions to bf16
  cvt_bf16_kernel<<<(HDIM * FDIM + 255) / 256, 256, 0, stream>>>(Wih, Wihb, HDIM * FDIM);
  cvt_bf16_kernel<<<(HDIM * HDIM + 255) / 256, 256, 0, stream>>>(Whh, Whhb, HDIM * HDIM);
  cvt_bf16_kernel<<<(ODIM * HDIM + 255) / 256, 256, 0, stream>>>(Who, Whob, ODIM * HDIM);

  // 2) depthwise causal conv -> x_f (bf16), tile staged to LDS by the TDM
  fracdiff_kernel<<<(TDIM / TT) * BDIM, 256, 0, stream>>>(x, kern, xf);

  // 3) input projection GEMM -> xp transposed [T][H][B], biases folded
  input_proj_kernel<<<(TDIM * BDIM) / 16, 256, 0, stream>>>(xf, Wihb, bih, bhh, xpT);

  // 4) sequential recurrence on one WGP + final output projection
  rnn_recurrence_kernel<<<1, 1024, 2 * BDIM * HDIM * 2, stream>>>(xpT, Whhb, Whob, bho, out);
}
